// EquivariantGNN_86191403696812
// MI455X (gfx1250) — compile-verified, matrix operand order checked by reading, not run on hardware
//
#include <hip/hip_runtime.h>

typedef _Float16 f16;
typedef __attribute__((ext_vector_type(16))) _Float16 v16h;
typedef __attribute__((ext_vector_type(8)))  _Float16 v8h;
typedef __attribute__((ext_vector_type(8)))  float    v8f;

#define HDIM  64
#define WAVES 8

// ---- helpers ---------------------------------------------------------------

// A-fragment (16-bit A-matrix 16x32 layout) from a contiguous 32-float slice:
// lane = M (lane&15), K half by lane>>4 (kb = (lane>>4)*8):
//   elements 0..7  = K = kb..kb+7, elements 8..15 = K = 16+kb..16+kb+7
__device__ __forceinline__ v16h load_a_f32(const float* __restrict__ base, int kb) {
  const float4* p = (const float4*)base;          // 32B aligned (kb in {0,8})
  float4 x0 = p[kb >> 2];
  float4 x1 = p[(kb >> 2) + 1];
  float4 x2 = p[4 + (kb >> 2)];
  float4 x3 = p[4 + (kb >> 2) + 1];
  v16h a;
  a[0] = (f16)x0.x; a[1] = (f16)x0.y; a[2]  = (f16)x0.z; a[3]  = (f16)x0.w;
  a[4] = (f16)x1.x; a[5] = (f16)x1.y; a[6]  = (f16)x1.z; a[7]  = (f16)x1.w;
  a[8] = (f16)x2.x; a[9] = (f16)x2.y; a[10] = (f16)x2.z; a[11] = (f16)x2.w;
  a[12]= (f16)x3.x; a[13]= (f16)x3.y; a[14] = (f16)x3.z; a[15] = (f16)x3.w;
  return a;
}

// Same fragment from an f16 row (two 16B loads, no converts).
__device__ __forceinline__ v16h load_a_f16(const f16* __restrict__ base, int kb) {
  v8h lo = *(const v8h*)(base + kb);        // 16B aligned (row stride 128B, kb*2 in {0,16})
  v8h hi = *(const v8h*)(base + 16 + kb);
  v16h a;
#pragma unroll
  for (int i = 0; i < 8; ++i) { a[i] = lo[i]; a[8 + i] = hi[i]; }
  return a;
}

// Stage a row-major [K,64] fp32 weight matrix into LDS as f16, pre-swizzled
// into B-matrix (32x16 f16) wave fragments: tile(s,nt) = 512 halfs, lane owns
// 16 contiguous halfs (one 32B ds_load at WMMA time).
__device__ __forceinline__ void stage_B(const float* __restrict__ W, f16* lds, int ksteps) {
  int total = ksteps * 4 * 512;
  for (int idx = threadIdx.x; idx < total; idx += blockDim.x) {
    int i    = idx & 15;
    int l    = (idx >> 4) & 31;
    int tile = idx >> 9;                  // tile = s*4 + nt
    int s    = tile >> 2;
    int nt   = tile & 3;
    int K = 32 * s + ((l >> 4) * 8) + (i & 7) + ((i & 8) ? 16 : 0);
    int n = nt * 16 + (l & 15);
    lds[idx] = (f16)W[K * HDIM + n];
  }
}

__device__ __forceinline__ v8f wmma_f16(v16h a, v16h b, v8f c) {
  return __builtin_amdgcn_wmma_f32_16x16x32_f16(false, a, false, b, (short)0, c, false, false);
}

// ---- kernels ---------------------------------------------------------------

__global__ void k_zero(float* __restrict__ p, int n) {
  int i = blockIdx.x * blockDim.x + threadIdx.x;
  if (i < n) p[i] = 0.0f;
}

// e0 = (f16)(h @ w_init + b_init)     ([N,32] @ [32,64])
__global__ __launch_bounds__(256) void k_init(const float* __restrict__ h,
                                              const float* __restrict__ w_init,
                                              const float* __restrict__ b_init,
                                              f16* __restrict__ out, int ntiles) {
  __shared__ __align__(32) f16 sB[4 * 512];
  stage_B(w_init, sB, 1);
  __syncthreads();
  int wave = threadIdx.x >> 5, lane = threadIdx.x & 31;
  int tile = blockIdx.x * WAVES + wave;
  if (tile >= ntiles) return;
  int m = lane & 15, hi = lane >> 4, kb = hi * 8;
  int node = tile * 16 + m;
  v16h a = load_a_f32(h + (size_t)node * 32, kb);
#pragma unroll
  for (int nt = 0; nt < 4; ++nt) {
    v16h b = *(const v16h*)&sB[nt * 512 + lane * 16];
    v8f c = {};
    c = wmma_f16(a, b, c);
    int nglob = nt * 16 + (lane & 15);
    float bias = b_init[nglob];
#pragma unroll
    for (int r = 0; r < 8; ++r) {
      int M = r + 8 * hi;
      out[(size_t)(tile * 16 + M) * HDIM + nglob] = (f16)(c[r] + bias);
    }
  }
}

// msg = relu(concat(embed[row], embed[col], dist) @ mw + mb); atomic scatter into aggr[col]
__global__ __launch_bounds__(256) void k_msg(const f16* __restrict__ embed,
                                             const float* __restrict__ xpos,
                                             const int* __restrict__ row,
                                             const int* __restrict__ col,
                                             const float* __restrict__ mw,
                                             const float* __restrict__ mb,
                                             float* __restrict__ aggr, int etiles) {
  __shared__ __align__(32) f16 sB[16 * 512];
  stage_B(mw, sB, 4);                       // first 128 rows of mw
  __syncthreads();
  int wave = threadIdx.x >> 5, lane = threadIdx.x & 31;
  int tile = blockIdx.x * WAVES + wave;
  if (tile >= etiles) return;
  int m = lane & 15, hi = lane >> 4, kb = hi * 8;
  int e = tile * 16 + m;
  int rs = row[e], cd = col[e];
  float dx = xpos[rs * 3 + 0] - xpos[cd * 3 + 0];
  float dy = xpos[rs * 3 + 1] - xpos[cd * 3 + 1];
  float dz = xpos[rs * 3 + 2] - xpos[cd * 3 + 2];
  float dist_m = dx * dx + dy * dy + dz * dz;

  v16h a0 = load_a_f16(embed + (size_t)rs * HDIM +  0, kb);
  v16h a1 = load_a_f16(embed + (size_t)rs * HDIM + 32, kb);
  v16h a2 = load_a_f16(embed + (size_t)cd * HDIM +  0, kb);
  v16h a3 = load_a_f16(embed + (size_t)cd * HDIM + 32, kb);

  float distR[8]; int colR[8];
#pragma unroll
  for (int r = 0; r < 8; ++r) {
    int M = r + 8 * hi;
    distR[r] = __shfl(dist_m, M, 32);
    colR[r]  = col[tile * 16 + M];
  }

#pragma unroll
  for (int nt = 0; nt < 4; ++nt) {
    v8f c = {};
#pragma unroll
    for (int s = 0; s < 4; ++s) {
      v16h b = *(const v16h*)&sB[(s * 4 + nt) * 512 + lane * 16];
      v16h a = (s == 0) ? a0 : (s == 1) ? a1 : (s == 2) ? a2 : a3;
      c = wmma_f16(a, b, c);
    }
    int nglob = nt * 16 + (lane & 15);
    float w128 = mw[128 * HDIM + nglob];    // dist column (rank-1 update)
    float bias = mb[nglob];
#pragma unroll
    for (int r = 0; r < 8; ++r) {
      float v = c[r] + distR[r] * w128 + bias;
      v = v > 0.0f ? v : 0.0f;
      atomicAdd(&aggr[(size_t)colR[r] * HDIM + nglob], v);
    }
  }
}

// out = embed @ rp + relu(concat(embed, aggr) @ uw + ub)
// OUT_T = f16 for the inter-layer buffer, float for the final output.
template <typename OUT_T>
__global__ __launch_bounds__(256) void k_update(const f16* __restrict__ embed,
                                                const float* __restrict__ aggr,
                                                const float* __restrict__ rp,
                                                const float* __restrict__ uw,
                                                const float* __restrict__ ub,
                                                OUT_T* __restrict__ out, int ntiles) {
  __shared__ __align__(32) f16 sU[16 * 512];
  __shared__ __align__(32) f16 sR[8 * 512];
  stage_B(uw, sU, 4);
  stage_B(rp, sR, 2);
  __syncthreads();
  int wave = threadIdx.x >> 5, lane = threadIdx.x & 31;
  int tile = blockIdx.x * WAVES + wave;
  if (tile >= ntiles) return;
  int m = lane & 15, hi = lane >> 4, kb = hi * 8;
  int node = tile * 16 + m;
  v16h a0 = load_a_f16(embed + (size_t)node * HDIM +  0, kb);
  v16h a1 = load_a_f16(embed + (size_t)node * HDIM + 32, kb);
  v16h a2 = load_a_f32(aggr  + (size_t)node * HDIM +  0, kb);
  v16h a3 = load_a_f32(aggr  + (size_t)node * HDIM + 32, kb);

#pragma unroll
  for (int nt = 0; nt < 4; ++nt) {
    v8f cu = {};
    v8f cr = {};
#pragma unroll
    for (int s = 0; s < 4; ++s) {
      v16h b = *(const v16h*)&sU[(s * 4 + nt) * 512 + lane * 16];
      v16h a = (s == 0) ? a0 : (s == 1) ? a1 : (s == 2) ? a2 : a3;
      cu = wmma_f16(a, b, cu);
    }
#pragma unroll
    for (int s = 0; s < 2; ++s) {
      v16h b = *(const v16h*)&sR[(s * 4 + nt) * 512 + lane * 16];
      v16h a = (s == 0) ? a0 : a1;
      cr = wmma_f16(a, b, cr);
    }
    int nglob = nt * 16 + (lane & 15);
    float bias = ub[nglob];
#pragma unroll
    for (int r = 0; r < 8; ++r) {
      int M = r + 8 * hi;
      float v = cu[r] + bias;
      v = v > 0.0f ? v : 0.0f;
      out[(size_t)(tile * 16 + M) * HDIM + nglob] = (OUT_T)(cr[r] + v);
    }
  }
}

// ---- launch ----------------------------------------------------------------

extern "C" void kernel_launch(void* const* d_in, const int* in_sizes, int n_in,
                              void* d_out, int out_size, void* d_ws, size_t ws_size,
                              hipStream_t stream) {
  (void)n_in; (void)out_size; (void)ws_size;
  const float* x      = (const float*)d_in[0];
  const float* h      = (const float*)d_in[1];
  const int*   eidx   = (const int*)  d_in[2];
  // d_in[3] = batch (unused)
  const float* w_init = (const float*)d_in[4];
  const float* b_init = (const float*)d_in[5];
  const float* rp0 = (const float*)d_in[6];
  const float* mw0 = (const float*)d_in[7];
  const float* mb0 = (const float*)d_in[8];
  const float* uw0 = (const float*)d_in[9];
  const float* ub0 = (const float*)d_in[10];
  const float* rp1 = (const float*)d_in[11];
  const float* mw1 = (const float*)d_in[12];
  const float* mb1 = (const float*)d_in[13];
  const float* uw1 = (const float*)d_in[14];
  const float* ub1 = (const float*)d_in[15];

  int N = in_sizes[1] / 32;        // h is [N, C_IN=32]
  int E = in_sizes[2] / 2;         // edge_index is [2, E]
  const int* row = eidx;
  const int* col = eidx + E;

  // workspace: e0 (f16), aggr (f32), e1 (f16)
  char* ws = (char*)d_ws;
  f16*   e0   = (f16*)ws;                                   ws += (size_t)N * HDIM * sizeof(f16);
  float* aggr = (float*)ws;                                 ws += (size_t)N * HDIM * sizeof(float);
  f16*   e1   = (f16*)ws;
  float* outp = (float*)d_out;

  int ntiles  = (N + 15) / 16;
  int etiles  = (E + 15) / 16;
  int nblocks = (ntiles + WAVES - 1) / WAVES;
  int eblocks = (etiles + WAVES - 1) / WAVES;
  int zblocks = (N * HDIM + 255) / 256;

  k_init<<<nblocks, 256, 0, stream>>>(h, w_init, b_init, e0, ntiles);

  k_zero<<<zblocks, 256, 0, stream>>>(aggr, N * HDIM);
  k_msg <<<eblocks, 256, 0, stream>>>(e0, x, row, col, mw0, mb0, aggr, etiles);
  k_update<f16><<<nblocks, 256, 0, stream>>>(e0, aggr, rp0, uw0, ub0, e1, ntiles);

  k_zero<<<zblocks, 256, 0, stream>>>(aggr, N * HDIM);
  k_msg <<<eblocks, 256, 0, stream>>>(e1, x, row, col, mw1, mb1, aggr, etiles);
  k_update<float><<<nblocks, 256, 0, stream>>>(e1, aggr, rp1, uw1, ub1, outp, ntiles);
}
